// dot_product_attention_28544352649216
// MI455X (gfx1250) — matrix operand order repaired and checked by reading, not compiled, not run on hardware
//
#include <hip/hip_runtime.h>

// ---------------------------------------------------------------------------
// causal attention: qkv = x @ W_qkv ; flash-attention over (q,k,v)
// B=8, T=2048, C=1024, HEAD=64.  bf16 WMMA compute, fp32 accumulate.
// ---------------------------------------------------------------------------

#define BATCH 8
#define SEQ   2048
#define CDIM  1024
#define HEAD  64
#define MROWS (BATCH * SEQ)   // 16384

typedef __attribute__((ext_vector_type(16))) __bf16          v16bf;
typedef __attribute__((ext_vector_type(8)))  float           v8f;
typedef __attribute__((ext_vector_type(16))) unsigned short  v16u;
typedef __attribute__((ext_vector_type(8)))  unsigned short  u8s;   // 16 B
typedef __attribute__((ext_vector_type(4)))  float           f4;    // 16 B
typedef __attribute__((ext_vector_type(4)))  int             v4i;   // 16 B

#define AS1 __attribute__((address_space(1)))
#define AS3 __attribute__((address_space(3)))

#if defined(__gfx1250__) && \
    __has_builtin(__builtin_amdgcn_global_load_async_to_lds_b128) && \
    __has_builtin(__builtin_amdgcn_s_wait_asynccnt)
#define HAVE_ASYNC_LDS 1
#else
#define HAVE_ASYNC_LDS 0
#endif

// fp32 -> bf16 via native conversion (v_cvt_pk_bf16_f32 when paired)
static __device__ __forceinline__ unsigned short f2bf(float f) {
  return __builtin_bit_cast(unsigned short, (__bf16)f);
}

static __device__ __forceinline__ v8f wmma_bf16(v16u a, v16u b, v8f c) {
  return __builtin_amdgcn_wmma_f32_16x16x32_bf16(
      false, __builtin_bit_cast(v16bf, a),
      false, __builtin_bit_cast(v16bf, b),
      (short)0, c, false, false);
}

// load 16 contiguous bf16 (two ds/global b128) as one WMMA operand half-pair
static __device__ __forceinline__ v16u ld16(const unsigned short* p) {
  union { v16u v; u8s h[2]; } u;
  u.h[0] = *(const u8s*)p;
  u.h[1] = *(const u8s*)(p + 8);
  return u.v;
}
// load two separated 8-element runs (A-layout operand)
static __device__ __forceinline__ v16u ld8x2(const unsigned short* p0,
                                             const unsigned short* p1) {
  union { v16u v; u8s h[2]; } u;
  u.h[0] = *(const u8s*)p0;
  u.h[1] = *(const u8s*)p1;
  return u.v;
}

// max / sum across the 16 lanes sharing a C/D-layout row
static __device__ __forceinline__ float rowmax16(float v) {
#pragma unroll
  for (int m = 1; m < 16; m <<= 1) v = fmaxf(v, __shfl_xor(v, m, 32));
  return v;
}
static __device__ __forceinline__ float rowsum16(float v) {
#pragma unroll
  for (int m = 1; m < 16; m <<= 1) v += __shfl_xor(v, m, 32);
  return v;
}

// ---------------------------------------------------------------------------
// Kernel 1: qkv projection. grid = (MROWS/64, 12), block = 128 (4 waves).
// ---------------------------------------------------------------------------
__global__ __launch_bounds__(128) void qkv_proj_kernel(
    const float* __restrict__ x, const float* __restrict__ W,
    unsigned short* __restrict__ Qb, unsigned short* __restrict__ Kb,
    unsigned short* __restrict__ Vb)
{
  __shared__ unsigned short Wt[16][CDIM + 8];

  const int tid = threadIdx.x;
  const int n0  = blockIdx.y * 16;            // panel of 16 output cols

  // stage W[:, n0:n0+16] transposed, bf16
  for (int i = tid; i < 16 * CDIM; i += 128) {
    int kk = i >> 4;
    int n  = i & 15;
    Wt[n][kk] = f2bf(W[(size_t)kk * 192 + n0 + n]);
  }
  __syncthreads();

  const int wave = tid >> 5;
  const int lane = tid & 31;
  const int lr   = lane & 15;
  const int half = lane >> 4;
  const int m0   = blockIdx.x * 64 + wave * 16;

  const float* xr = x + (size_t)(m0 + lr) * CDIM;

  v8f acc = {};
  for (int ks = 0; ks < CDIM; ks += 32) {
    // A operand: 16x32 tile of x (A-layout), converted to bf16
    const f4* xp = (const f4*)(xr + ks + half * 8);        // K j = 0..7
    const f4* xq = (const f4*)(xr + ks + 16 + half * 8);   // K j = 8..15
    f4 a0 = xp[0], a1 = xp[1], a2 = xq[0], a3 = xq[1];
    v16u au;
#pragma unroll
    for (int j = 0; j < 4; ++j) {
      au[j]      = f2bf(a0[j]);
      au[4 + j]  = f2bf(a1[j]);
      au[8 + j]  = f2bf(a2[j]);
      au[12 + j] = f2bf(a3[j]);
    }
    // B operand: 32x16 tile of W (B-layout) from LDS
    v16u bu = ld8x2(&Wt[lr][ks + half * 8], &Wt[lr][ks + 16 + half * 8]);

    acc = wmma_bf16(au, bu, acc);
  }

  // scatter result tile into Q/K/V bf16 workspace
  unsigned short* dst = (n0 < 64) ? Qb : (n0 < 128 ? Kb : Vb);
  const int col = (n0 & 63) + lr;
#pragma unroll
  for (int i = 0; i < 8; ++i) {
    int m = m0 + half * 8 + i;
    dst[(size_t)m * HEAD + col] = f2bf(acc[i]);
  }
}

// ---------------------------------------------------------------------------
// Kernel 2: causal flash attention. grid = (SEQ/128, BATCH), block = 256.
// ---------------------------------------------------------------------------
__global__ __launch_bounds__(256) void flash_attn_kernel(
    const unsigned short* __restrict__ Qb,
    const unsigned short* __restrict__ Kb,
    const unsigned short* __restrict__ Vb,
    float* __restrict__ out)
{
  __shared__ unsigned short Ks[64][HEAD + 8];   // K chunk, row stride 144 B
  __shared__ unsigned short Vt[HEAD][64 + 8];   // V chunk transposed
  __shared__ unsigned short Ps[8][16][32 + 8];  // per-wave P staging

  const int tid  = threadIdx.x;
  const int wave = tid >> 5;
  const int lane = tid & 31;
  const int lr   = lane & 15;
  const int half = lane >> 4;
  const int b    = blockIdx.y;
  const int q0   = blockIdx.x * 128 + wave * 16;
  const size_t base = (size_t)b * SEQ * HEAD;

  // Q operand (A-layout), head dims [0,32) and [32,64)
  const unsigned short* qr = Qb + base + (size_t)(q0 + lr) * HEAD;
  v16u aq0 = ld8x2(qr + half * 8,      qr + 16 + half * 8);
  v16u aq1 = ld8x2(qr + 32 + half * 8, qr + 48 + half * 8);

  v8f   accO[4] = {};
  float mrow[8], lrow[8];
#pragma unroll
  for (int i = 0; i < 8; ++i) { mrow[i] = -1e30f; lrow[i] = 0.f; }

  const int r   = tid >> 2;          // 0..63: staged key row
  const int c16 = (tid & 3) * 16;    // 16-col segment

  const int kv_hi = blockIdx.x * 128 + 128;     // causal bound for this block
  for (int kv0 = 0; kv0 < kv_hi; kv0 += 64) {
    // ---- stage 64 keys of K (async if available) and V^T ----
    {
      const unsigned short* kr = Kb + base + (size_t)(kv0 + r) * HEAD + c16;
      const unsigned short* vr = Vb + base + (size_t)(kv0 + r) * HEAD + c16;
#if HAVE_ASYNC_LDS
      __builtin_amdgcn_global_load_async_to_lds_b128(
          (AS1 v4i*)const_cast<unsigned short*>(kr),
          (AS3 v4i*)&Ks[r][c16], 0, 0);
      __builtin_amdgcn_global_load_async_to_lds_b128(
          (AS1 v4i*)const_cast<unsigned short*>(kr),
          (AS3 v4i*)&Ks[r][c16], 16, 0);
#else
      *(u8s*)&Ks[r][c16]     = *(const u8s*)kr;
      *(u8s*)&Ks[r][c16 + 8] = *(const u8s*)(kr + 8);
#endif
      union { u8s h[2]; unsigned short s[16]; } vv;
      vv.h[0] = *(const u8s*)vr;
      vv.h[1] = *(const u8s*)(vr + 8);
#pragma unroll
      for (int j = 0; j < 16; ++j) Vt[c16 + j][r] = vv.s[j];
    }
#if HAVE_ASYNC_LDS
    __builtin_amdgcn_s_wait_asynccnt(0);
#endif
    __syncthreads();

    for (int kk = 0; kk < 64; kk += 32) {
      const int ksg = kv0 + kk;                  // first key of sub-block
      if (ksg <= q0) {                           // causal: any key usable?
        // ---- S = Q K^T for two 16-key tiles ----
        v8f s[2];
#pragma unroll
        for (int t = 0; t < 2; ++t) {
          const unsigned short* krow = &Ks[kk + t * 16 + lr][0];
          v16u b0 = ld8x2(krow + half * 8, krow + 16 + half * 8);
          v16u b1 = ld8x2(krow + 32 + half * 8, krow + 48 + half * 8);
          v8f acc = {};
          acc  = wmma_bf16(aq0, b0, acc);
          s[t] = wmma_bf16(aq1, b1, acc);
        }

        // ---- scale, causal mask, online softmax over 16x32 strip ----
#pragma unroll
        for (int i = 0; i < 8; ++i) {
          const int ig = q0 + half * 8 + i;
          float s0 = s[0][i] * 0.125f;
          float s1 = s[1][i] * 0.125f;
          if (ksg + lr      > ig) s0 = -1e30f;
          if (ksg + 16 + lr > ig) s1 = -1e30f;
          float tm = fmaxf(rowmax16(s0), rowmax16(s1));
          float nm = fmaxf(mrow[i], tm);
          float c  = __expf(mrow[i] - nm);
          float p0 = __expf(s0 - nm);
          float p1 = __expf(s1 - nm);
          lrow[i] = lrow[i] * c + rowsum16(p0) + rowsum16(p1);
          mrow[i] = nm;
          accO[0][i] *= c; accO[1][i] *= c;
          accO[2][i] *= c; accO[3][i] *= c;
          Ps[wave][half * 8 + i][lr]      = f2bf(p0);
          Ps[wave][half * 8 + i][16 + lr] = f2bf(p1);
        }
        asm volatile("" ::: "memory");  // order Ps stores before reloads

        // ---- P (A-layout) from LDS ----
        const unsigned short* pr = &Ps[wave][lr][0];
        v16u ap = ld8x2(pr + half * 8, pr + 16 + half * 8);

        // ---- O += P V over four 16-wide head tiles ----
#pragma unroll
        for (int nt = 0; nt < 4; ++nt) {
          v16u bv = ld8x2(&Vt[nt * 16 + lr][kk + half * 8], &Vt[nt * 16 + lr][kk + 16 + half * 8]);
          accO[nt] = wmma_bf16(ap, bv, accO[nt]);
        }
        asm volatile("" ::: "memory");
      }
    }
    __syncthreads();
  }

  // ---- epilogue: out = O / l ----
#pragma unroll
  for (int i = 0; i < 8; ++i) {
    float inv = 1.0f / lrow[i];
    float* op = out + base + (size_t)(q0 + half * 8 + i) * HEAD;
    op[lr]      = accO[0][i] * inv;
    op[16 + lr] = accO[1][i] * inv;
    op[32 + lr] = accO[2][i] * inv;
    op[48 + lr] = accO[3][i] * inv;
  }
}

// ---------------------------------------------------------------------------
extern "C" void kernel_launch(void* const* d_in, const int* in_sizes, int n_in,
                              void* d_out, int out_size, void* d_ws, size_t ws_size,
                              hipStream_t stream) {
  const float* x = (const float*)d_in[0];   // [B,T,C] fp32
  const float* W = (const float*)d_in[1];   // [C, 3*HEAD] fp32
  float* out     = (float*)d_out;           // [B,T,HEAD] fp32

  unsigned short* ws = (unsigned short*)d_ws;
  unsigned short* Qb = ws;                                   // [MROWS,64] bf16
  unsigned short* Kb = ws + (size_t)MROWS * HEAD;            // [MROWS,64] bf16
  unsigned short* Vb = ws + (size_t)2 * MROWS * HEAD;        // [MROWS,64] bf16

  qkv_proj_kernel<<<dim3(MROWS / 64, 12), 128, 0, stream>>>(x, W, Qb, Kb, Vb);
  flash_attn_kernel<<<dim3(SEQ / 128, BATCH), 256, 0, stream>>>(Qb, Kb, Vb, out);
}
